// KNNMemory_6597069766776
// MI455X (gfx1250) — matrix operand order repaired and checked by reading, not compiled, not run on hardware
//
#include <hip/hip_runtime.h>
#include <math.h>

typedef __attribute__((ext_vector_type(16))) _Float16 v16h;
typedef __attribute__((ext_vector_type(8)))  _Float16 v8h;
typedef __attribute__((ext_vector_type(8)))  float    v8f;

#define H_ 16
#define Q_ 1024
#define M_ 32768
#define D_ 64
#define K_ 32
#define NWAVES 8
#define NTILES (M_ / 16)          // 2048 key tiles of 16
#define QT_ 32                    // queries per block (2 sub-tiles of 16)

__device__ __forceinline__ float wave_sumsq(float v) {
  #pragma unroll
  for (int off = 16; off > 0; off >>= 1)
    v += __shfl_xor(v, off, 32);
  return v;
}

// Normalize rows of length D=64: x / (||x|| + eps), emit f16.
__global__ __launch_bounds__(256)
void normalize_rows_f16(const float* __restrict__ src,
                        _Float16* __restrict__ dst, int nrows) {
  const int gwave = (int)((blockIdx.x * blockDim.x + threadIdx.x) >> 5);
  const int lane  = threadIdx.x & 31;
  if (gwave >= nrows) return;
  const float2 v = ((const float2*)(src + (size_t)gwave * D_))[lane];
  const float ss  = wave_sumsq(v.x * v.x + v.y * v.y);
  const float inv = 1.0f / (sqrtf(ss) + 1e-8f);
  union { _Float16 h[2]; unsigned u; } pk;
  pk.h[0] = (_Float16)(v.x * inv);
  pk.h[1] = (_Float16)(v.y * inv);
  ((unsigned*)dst)[(size_t)gwave * (D_ / 2) + lane] = pk.u;
}

// A operand (16 keys x 32 dims): K-chunk base c32 (0/32), lane half hb (0/8).
__device__ __forceinline__ v16h load_a(const _Float16* krow, int c32, int hb) {
  v16h a;
  const v8h lo = *(const v8h*)(krow + c32 + 2 * hb);
  const v8h hi = *(const v8h*)(krow + c32 + 2 * hb + 8);
  #pragma unroll
  for (int i = 0; i < 8; ++i) { a[i] = lo[i]; a[i + 8] = hi[i]; }
  return a;
}

// Insert up to 8 candidates (one lane's C-tile column) into this thread's
// private sorted list, with register threshold + tile-level quick reject.
__device__ __forceinline__ void insert8(float (* __restrict__ sp)[256],
                                        int   (* __restrict__ ip)[256],
                                        int tid, float& thr,
                                        const v8f& acc, int base) {
  float mx = acc[0];
  #pragma unroll
  for (int v = 1; v < 8; ++v) mx = fmaxf(mx, acc[v]);
  if (mx * 0.125f > thr) {
    #pragma unroll
    for (int v = 0; v < 8; ++v) {
      const float s = acc[v] * 0.125f;              // 1/sqrt(64)
      if (s > thr) {
        const int idx = base + v;
        int j = K_ - 1;
        #pragma clang loop unroll(disable)
        for (; j > 0; --j) {
          const float t = sp[j - 1][tid];
          if (t >= s) break;
          sp[j][tid] = t;
          ip[j][tid] = ip[j - 1][tid];
        }
        sp[j][tid] = s;
        ip[j][tid] = idx;
        thr = sp[K_ - 1][tid];                      // refresh threshold
      }
    }
  }
}

// One workgroup per (head, 32-query tile). 8 waves stripe over key tiles.
// Per key tile: one A load feeds TWO query sub-tiles (4 interleaved WMMAs),
// halving L2 key traffic vs 16-query blocks and filling WMMA hazard slots.
__global__ __launch_bounds__(256)
void knn_topk_kernel(const _Float16* __restrict__ qn,
                     const _Float16* __restrict__ kn,
                     const float* __restrict__ values,
                     float* __restrict__ out) {
  __shared__ float topS[2][K_][256];   // 64 KB  (plane = query sub-tile)
  __shared__ int   topI[2][K_][256];   // 64 KB
  __shared__ int   finalIdx[QT_][K_];  //  4 KB

  const int tid  = threadIdx.x;
  const int wave = tid >> 5;
  const int lane = tid & 31;
  const int h    = blockIdx.x >> 5;          // Q_/QT_ == 32 tiles per head
  const int q0   = (blockIdx.x & 31) << 5;

  #pragma unroll
  for (int j = 0; j < K_; ++j) {
    topS[0][j][tid] = -1e30f; topI[0][j][tid] = 0;
    topS[1][j][tid] = -1e30f; topI[1][j][tid] = 0;
  }

  // ---- B operands: two 16-query sub-tiles, resident for whole block ----
  const int qcol = lane & 15;
  const int kb   = (lane >> 4) << 4;
  const _Float16* qrow0 = qn + ((size_t)h * Q_ + q0 + qcol) * D_;
  const _Float16* qrow1 = qrow0 + 16 * D_;
  const v16h b0 = *(const v16h*)(qrow0 + kb);
  const v16h b1 = *(const v16h*)(qrow0 + 32 + kb);
  const v16h b2 = *(const v16h*)(qrow1 + kb);
  const v16h b3 = *(const v16h*)(qrow1 + 32 + kb);

  const _Float16* kh = kn + (size_t)h * M_ * D_;
  const int hb   = (lane >> 4) << 3;   // A K-half offset: 0 or 8
  const int arow = lane & 15;

  float thr0 = -1e30f, thr1 = -1e30f;

  // Software pipeline: preload tile `wave`, compute while loading next.
  const _Float16* krow = kh + (size_t)(wave * 16 + arow) * D_;
  v16h a0 = load_a(krow, 0,  hb);
  v16h a1 = load_a(krow, 32, hb);

  for (int nt = wave; nt < NTILES; nt += NWAVES) {
    const int m0 = nt << 4;

    v16h na0, na1;
    const int nnt = nt + NWAVES;
    if (nnt < NTILES) {
      const _Float16* nkrow = kh + (size_t)(nnt * 16 + arow) * D_;
      __builtin_prefetch(nkrow + (size_t)NWAVES * 16 * D_, 0, 3);
      na0 = load_a(nkrow, 0,  hb);
      na1 = load_a(nkrow, 32, hb);
    }

    // Interleave the two accumulators: independent back-to-back WMMAs.
    v8f acc0 = {}, acc1 = {};
    acc0 = __builtin_amdgcn_wmma_f32_16x16x32_f16(false, a0, false, b0,
                                                  (short)0, acc0, false, false);
    acc1 = __builtin_amdgcn_wmma_f32_16x16x32_f16(false, a0, false, b2,
                                                  (short)0, acc1, false, false);
    acc0 = __builtin_amdgcn_wmma_f32_16x16x32_f16(false, a1, false, b1,
                                                  (short)0, acc0, false, false);
    acc1 = __builtin_amdgcn_wmma_f32_16x16x32_f16(false, a1, false, b3,
                                                  (short)0, acc1, false, false);
    a0 = na0; a1 = na1;

    const int base = m0 + hb;
    insert8(topS[0], topI[0], tid, thr0, acc0, base);
    insert8(topS[1], topI[1], tid, thr1, acc1, base);
  }
  __syncthreads();

  // ---- merge: one full wave, lane = query-in-block (t = lane>>4 plane,
  //      q = lane&15 column); 16 sorted source lists per query ----
  if (wave == 0) {
    const int t = lane >> 4;
    const int q = lane & 15;
    int p[16];
    #pragma unroll
    for (int i = 0; i < 16; ++i) p[i] = 0;
    for (int k = 0; k < K_; ++k) {
      float best = -1e30f; int bi = 0;
      #pragma unroll
      for (int i = 0; i < 16; ++i) {
        const int lid = ((i >> 1) << 5) + ((i & 1) << 4) + q;
        const float s = (p[i] < K_) ? topS[t][p[i]][lid] : -1e30f;
        if (s > best) { best = s; bi = i; }
      }
      const int lid = ((bi >> 1) << 5) + ((bi & 1) << 4) + q;
      const int idx = topI[t][p[bi]][lid];
      p[bi]++;
      const int ql = (t << 4) + q;                 // query within block
      finalIdx[ql][k] = idx;
      const size_t o = ((size_t)h * Q_ + q0 + ql) * K_ + k;
      out[o] = best;                                      // scores (f32)
      ((int*)out)[(size_t)H_ * Q_ * K_ + o] = idx;        // indices (i32)
    }
  }
  __syncthreads();

  // ---- gather values [32 q][32 k][64 d] with float4 ----
  float* gout = out + (size_t)2 * H_ * Q_ * K_;
  const float* vh = values + (size_t)h * M_ * D_;
  for (int e = tid; e < QT_ * K_ * (D_ / 4); e += 256) {
    const int d4   = e & 15;
    const int pair = e >> 4;
    const int k    = pair & (K_ - 1);
    const int q    = pair >> 5;
    const int idx  = finalIdx[q][k];
    const float4 val = ((const float4*)(vh + (size_t)idx * D_))[d4];
    float4* dst = (float4*)(gout + (((size_t)h * Q_ + q0 + q) * K_ + k) * D_);
    dst[d4] = val;
  }
}

extern "C" void kernel_launch(void* const* d_in, const int* in_sizes, int n_in,
                              void* d_out, int out_size, void* d_ws, size_t ws_size,
                              hipStream_t stream) {
  const float* queries = (const float*)d_in[0];
  const float* keys    = (const float*)d_in[1];
  const float* values  = (const float*)d_in[2];
  float* out = (float*)d_out;

  _Float16* qn = (_Float16*)d_ws;                       // 2 MB
  _Float16* kn = qn + (size_t)H_ * Q_ * D_;             // 64 MB

  {
    const int nrows = H_ * Q_;
    normalize_rows_f16<<<(nrows + 7) / 8, 256, 0, stream>>>(queries, qn, nrows);
  }
  {
    const int nrows = H_ * M_;
    normalize_rows_f16<<<(nrows + 7) / 8, 256, 0, stream>>>(keys, kn, nrows);
  }
  knn_topk_kernel<<<H_ * (Q_ / QT_), 256, 0, stream>>>(qn, kn, values, out);
}
